// GCN_62371515072690
// MI455X (gfx1250) — matrix operand order repaired and checked by reading, not compiled, not run on hardware
//
#include <hip/hip_runtime.h>

typedef __bf16 bf16;
typedef __attribute__((ext_vector_type(16))) __bf16 bf16x16;
typedef __attribute__((ext_vector_type(8)))  __bf16 bf16x8;
typedef __attribute__((ext_vector_type(8)))  float   v8f;
typedef __attribute__((ext_vector_type(4)))  unsigned int u32x4;

union FragAB { u32x4 q[2]; bf16x16 v; };
union HalfD  { u32x4 q;    bf16x8  v; };

// ---------------------------------------------------------------------------
// Generic bf16 WMMA GEMM.
//   math:   D[M,N] = A[M,K] * B[K,N]  (+bias[N], optional relu)
//   A storage: row-major [M][K]   (lda = K stride)
//   B storage: [N][K]  (i.e. B^T row-major; ldb = K stride)
//   D storage: transposed [N][M]  (ldc = M stride), bf16
// One wave computes a 64x64 tile via 4x4 v_wmma_f32_16x16x32_bf16.
// K multiple of 32; M,N multiples of 16. Out-of-range 16-blocks are CLAMPED on
// load (branch-free; garbage accumulators are never stored) and guarded on
// store. Hand-pipelined, unroll-by-2 K loop with two statically-named fragment
// sets (a0/b0, a1/b1) so everything stays in VGPRs (no runtime-indexed arrays
// -> no scratch spills).
// ---------------------------------------------------------------------------
__global__ __launch_bounds__(32)
void wmma_gemm(const bf16* __restrict__ A, int lda, long aBatch,
               const bf16* __restrict__ Bm, int ldb, long bBatch,
               bf16* __restrict__ C, int ldc, long cBatch,
               int M, int N, int K,
               const float* __restrict__ bias, int relu)
{
    const int lane = threadIdx.x;       // wave32
    const int h    = lane >> 4;         // 0 or 1
    const int l16  = lane & 15;
    const int tm   = blockIdx.x * 64;
    const int tn   = blockIdx.y * 64;
    const int b    = blockIdx.z;

    A  += (long)b * aBatch;
    Bm += (long)b * bBatch;
    C  += (long)b * cBatch;

    bool mok[4], nok[4];
    const bf16* aptr[4];
    const bf16* bptr[4];
#pragma unroll
    for (int i = 0; i < 4; ++i) {
        mok[i] = (tm + 16 * i) < M;
        int mrow = mok[i] ? (tm + 16 * i) : (M - 16);   // clamp: always valid memory
        // 16-bit A 16x32 layout: lanes 0-15 K[0..7]/K[16..23], lanes 16-31 K[8..15]/K[24..31]
        aptr[i] = A + (long)(mrow + l16) * lda + h * 8;
    }
#pragma unroll
    for (int j = 0; j < 4; ++j) {
        nok[j] = (tn + 16 * j) < N;
        int nrow = nok[j] ? (tn + 16 * j) : (N - 16);
        // B 32x16 layout: lane holds column n; lanes 0-15 K[0..7]/K[16..23], lanes 16-31 K[8..15]/K[24..31]
        bptr[j] = Bm + (long)(nrow + l16) * ldb + h * 8;
    }

    v8f acc[4][4];
#pragma unroll
    for (int i = 0; i < 4; ++i)
#pragma unroll
        for (int j = 0; j < 4; ++j)
#pragma unroll
            for (int r = 0; r < 8; ++r) acc[i][j][r] = 0.0f;

    FragAB a0[4], b0[4], a1[4], b1[4];

    auto load_slab = [&](FragAB (&fa)[4], FragAB (&fb)[4], int kk) {
#pragma unroll
        for (int i = 0; i < 4; ++i) {
            fa[i].q[0] = *(const u32x4*)(aptr[i] + kk);
            fa[i].q[1] = *(const u32x4*)(aptr[i] + kk + 16);
        }
#pragma unroll
        for (int j = 0; j < 4; ++j) {
            fb[j].q[0] = *(const u32x4*)(bptr[j] + kk);
            fb[j].q[1] = *(const u32x4*)(bptr[j] + kk + 16);
        }
    };
    auto mma = [&](FragAB (&fa)[4], FragAB (&fb)[4]) {
#pragma unroll
        for (int i = 0; i < 4; ++i)
#pragma unroll
            for (int j = 0; j < 4; ++j)
                acc[i][j] = __builtin_amdgcn_wmma_f32_16x16x32_bf16(
                    false, fa[i].v, false, fb[j].v, (short)0, acc[i][j], false, false);
    };

    const int nk = K >> 5;              // K / 32, K always a multiple of 32
    load_slab(a0, b0, 0);
    int t = 0;
    for (; t + 1 < nk; t += 2) {
        const int k1 = (t + 1) << 5;
        const int k2 = ((t + 2 < nk) ? (t + 2) : t) << 5;   // clamp (redundant reload)
        load_slab(a1, b1, k1);
        mma(a0, b0);
        load_slab(a0, b0, k2);
        mma(a1, b1);
    }
    if (t < nk) mma(a0, b0);            // odd-nk tail (slab already resident)

    // D layout: lane l holds column N = l%16, rows M = 8*(l/16) + r, r=0..7 (contiguous)
#pragma unroll
    for (int j = 0; j < 4; ++j) {
        if (!nok[j]) continue;
        const int n = tn + 16 * j + l16;
        const float bs = bias ? bias[n] : 0.0f;
#pragma unroll
        for (int i = 0; i < 4; ++i) {
            if (!mok[i]) continue;
            HalfD o;
#pragma unroll
            for (int r = 0; r < 8; ++r) {
                float v = acc[i][j][r] + bs;
                if (relu) v = v > 0.0f ? v : 0.0f;
                o.v[r] = (bf16)v;
            }
            *(bf16x8*)(C + (long)n * ldc + tm + 16 * i + 8 * h) = o.v;
        }
    }
}

// ---------------------------------------------------------------------------
// Small helper kernels (f32 scalar; negligible FLOPs)
// ---------------------------------------------------------------------------
__global__ void k_zero_bf16(bf16* p, long n) {
    long i = (long)blockIdx.x * blockDim.x + threadIdx.x;
    if (i < n) p[i] = (bf16)0.0f;
}

__global__ void k_deg(const float* __restrict__ adj, float* __restrict__ d, int Nn) {
    int n = blockIdx.x * blockDim.x + threadIdx.x;
    if (n >= Nn) return;
    const float* row = adj + (long)n * Nn;
    float s = 1.0f;                    // +I self loop
    for (int m = 0; m < Nn; ++m) s += row[m];
    d[n] = rsqrtf(s);
}

__global__ void k_ahat(const float* __restrict__ adj, const float* __restrict__ d,
                       bf16* __restrict__ Ah, int Nn, int ldn) {
    long idx = (long)blockIdx.x * blockDim.x + threadIdx.x;
    long total = (long)Nn * ldn;
    if (idx >= total) return;
    int m = (int)(idx / ldn), n = (int)(idx % ldn);
    float v = 0.0f;
    if (n < Nn) {
        float a = adj[(long)m * Nn + n] + (m == n ? 1.0f : 0.0f);
        v = d[m] * d[n] * a;
    }
    Ah[idx] = (bf16)v;
}

// W^T with zero-padded K: out[o][kp] = kp<K ? w[kp][o] : 0
__global__ void k_wtr(const float* __restrict__ w, bf16* __restrict__ out,
                      int K, int O, int ldk) {
    int idx = blockIdx.x * blockDim.x + threadIdx.x;
    if (idx >= O * ldk) return;
    int kp = idx % ldk, o = idx / ldk;
    out[(long)o * ldk + kp] = (kp < K) ? (bf16)w[(long)kp * O + o] : (bf16)0.0f;
}

// zero pad columns n in [Nn, ldn) of feature-major tensor T[b][f][n]
__global__ void k_zpad(bf16* T, int Fdim, long tb, int Nn, int ldn, int total) {
    int idx = blockIdx.x * blockDim.x + threadIdx.x;
    if (idx >= total) return;
    int pad = ldn - Nn;
    int i = idx % pad;
    int f = (idx / pad) % Fdim;
    int b = idx / (pad * Fdim);
    T[(long)b * tb + (long)f * ldn + Nn + i] = (bf16)0.0f;
}

// scatter: T0[b][f][sid(b,j)] = feat(b,j,f) for j < S-1
__global__ void k_scatter(const float* __restrict__ x, bf16* __restrict__ T0,
                          int B, int S, int Fp1, int ldn, long tb) {
    int idx = blockIdx.x * blockDim.x + threadIdx.x;
    int F = Fp1 - 1;                     // 16
    if (idx >= B * (S - 1) * F) return;
    int f = idx % F;
    int j = (idx / F) % (S - 1);
    int b = idx / (F * (S - 1));
    const float* row = x + ((long)b * S + j) * Fp1;
    int sid = (int)row[Fp1 - 2];
    float v = (f < Fp1 - 2) ? row[f] : row[Fp1 - 1];
    T0[(long)b * tb + (long)f * ldn + sid] = (bf16)v;
}

// gather: g[b][f] = T3[b][f][sid_last(b)]
__global__ void k_gather(const float* __restrict__ x, const bf16* __restrict__ T3,
                         float* __restrict__ g, int B, int S, int Fp1,
                         int GH, int ldn, long tb) {
    int idx = blockIdx.x * blockDim.x + threadIdx.x;
    if (idx >= B * GH) return;
    int f = idx % GH, b = idx / GH;
    int sid = (int)x[((long)b * S + (S - 1)) * Fp1 + (Fp1 - 2)];
    g[idx] = (float)T3[(long)b * tb + (long)f * ldn + sid];
}

// fc0 pre-activation: out[b][o] = x[b,-1,:15] @ w + bias
__global__ void k_fc0(const float* __restrict__ x, const float* __restrict__ w,
                      const float* __restrict__ bias, float* __restrict__ out,
                      int B, int S, int Fp1, int O) {
    int idx = blockIdx.x * blockDim.x + threadIdx.x;
    if (idx >= B * O) return;
    int o = idx % O, b = idx / O;
    const float* row = x + ((long)b * S + (S - 1)) * Fp1;
    float s = bias[o];
    for (int k = 0; k < Fp1 - 2; ++k) s += row[k] * w[(long)k * O + o];
    out[idx] = s;
}

// batchnorm (biased var over batch) + leaky relu(0.01), one thread per column
__global__ void k_bn_leaky(const float* __restrict__ in, const float* __restrict__ gamma,
                           const float* __restrict__ beta, float* __restrict__ out,
                           int B, int O) {
    int o = blockIdx.x * blockDim.x + threadIdx.x;
    if (o >= O) return;
    float mu = 0.0f, sq = 0.0f;
    for (int b = 0; b < B; ++b) { float v = in[(long)b * O + o]; mu += v; sq += v * v; }
    mu /= (float)B;
    float var = sq / (float)B - mu * mu;
    float inv = rsqrtf(var + 1e-5f);
    float gm = gamma[o], be = beta[o];
    for (int b = 0; b < B; ++b) {
        float t = gm * (in[(long)b * O + o] - mu) * inv + be;
        out[(long)b * O + o] = t >= 0.0f ? t : 0.01f * t;
    }
}

// fc1 pre-activation over concat(h0, g): out[b][o] = [h0,g][b,:] @ w + bias
__global__ void k_fc1(const float* __restrict__ h0, const float* __restrict__ g,
                      const float* __restrict__ w, const float* __restrict__ bias,
                      float* __restrict__ out, int B, int FH, int GH, int O) {
    int idx = blockIdx.x * blockDim.x + threadIdx.x;
    if (idx >= B * O) return;
    int o = idx % O, b = idx / O;
    float s = bias[o];
    for (int k = 0; k < FH; ++k) s += h0[(long)b * FH + k] * w[(long)k * O + o];
    for (int k = 0; k < GH; ++k) s += g[(long)b * GH + k] * w[(long)(FH + k) * O + o];
    out[idx] = s;
}

__global__ void k_head(const float* __restrict__ h1, const float* __restrict__ w2,
                       const float* __restrict__ b2, float* __restrict__ out,
                       int B, int O) {
    int b = blockIdx.x * blockDim.x + threadIdx.x;
    if (b >= B) return;
    float s = b2[0];
    for (int o = 0; o < O; ++o) s += h1[(long)b * O + o] * w2[o];
    out[b] = 1.0f / (1.0f + expf(-s));
}

// ---------------------------------------------------------------------------
extern "C" void kernel_launch(void* const* d_in, const int* in_sizes, int n_in,
                              void* d_out, int out_size, void* d_ws, size_t ws_size,
                              hipStream_t stream)
{
    (void)in_sizes; (void)n_in; (void)out_size; (void)ws_size;
    const float* x      = (const float*)d_in[0];
    const float* adj    = (const float*)d_in[1];
    const float* w_gc1  = (const float*)d_in[2];
    const float* b_gc1  = (const float*)d_in[3];
    const float* w_gc2  = (const float*)d_in[4];
    const float* b_gc2  = (const float*)d_in[5];
    const float* w_gc3  = (const float*)d_in[6];
    const float* b_gc3  = (const float*)d_in[7];
    const float* w_fc0  = (const float*)d_in[8];
    const float* b_fc0  = (const float*)d_in[9];
    const float* g_fc0  = (const float*)d_in[10];
    const float* be_fc0 = (const float*)d_in[11];
    const float* w_fc1  = (const float*)d_in[12];
    const float* b_fc1  = (const float*)d_in[13];
    const float* g_fc1  = (const float*)d_in[14];
    const float* be_fc1 = (const float*)d_in[15];
    const float* w_fc2  = (const float*)d_in[16];
    const float* b_fc2  = (const float*)d_in[17];

    constexpr int B = 128, Nn = 2000, LDN = 2016, S = 64, Fp1 = 17;
    constexpr int GH2 = 256, GH = 512, FH = 512, K1 = 32;

    size_t off = 0;
    auto carve = [&](size_t bytes) -> void* {
        void* p = (char*)d_ws + off;
        off += (bytes + 255) & ~(size_t)255;
        return p;
    };
    float* dvec = (float*)carve((size_t)Nn * 4);
    bf16*  Ahat = (bf16*) carve((size_t)Nn * LDN * 2);
    bf16*  W1T  = (bf16*) carve((size_t)GH2 * K1 * 2);
    bf16*  W2T  = (bf16*) carve((size_t)GH2 * GH2 * 2);
    bf16*  W3T  = (bf16*) carve((size_t)GH * GH2 * 2);
    bf16*  T0   = (bf16*) carve((size_t)B * 16 * LDN * 2);
    bf16*  Z1   = (bf16*) carve((size_t)B * Nn * K1 * 2);
    bf16*  T1   = (bf16*) carve((size_t)B * GH2 * LDN * 2);
    bf16*  Z23  = (bf16*) carve((size_t)B * Nn * GH2 * 2);   // Z2 then Z3 (sequential reuse)
    bf16*  T2   = (bf16*) carve((size_t)B * GH2 * LDN * 2);
    bf16*  T3   = (bf16*) carve((size_t)B * GH * LDN * 2);
    float* h0p  = (float*)carve((size_t)B * FH * 4);
    float* h0   = (float*)carve((size_t)B * FH * 4);
    float* gsel = (float*)carve((size_t)B * GH * 4);
    float* h1p  = (float*)carve((size_t)B * FH * 4);
    float* h1   = (float*)carve((size_t)B * FH * 4);

    auto blk = [](long n) { return (unsigned)((n + 255) / 256); };

    // --- prep: degrees, Ahat(bf16, padded), transposed weights ---
    k_deg <<<blk(Nn), 256, 0, stream>>>(adj, dvec, Nn);
    k_ahat<<<blk((long)Nn * LDN), 256, 0, stream>>>(adj, dvec, Ahat, Nn, LDN);
    k_wtr <<<blk((long)GH2 * K1), 256, 0, stream>>>(w_gc1, W1T, 16, GH2, K1);
    k_wtr <<<blk((long)GH2 * GH2), 256, 0, stream>>>(w_gc2, W2T, GH2, GH2, GH2);
    k_wtr <<<blk((long)GH * GH2), 256, 0, stream>>>(w_gc3, W3T, GH2, GH, GH2);

    // --- zero-init buffers whose pads/holes are read by WMMA ---
    k_zero_bf16<<<blk((long)B * 16 * LDN), 256, 0, stream>>>(T0, (long)B * 16 * LDN);
    k_zero_bf16<<<blk((long)B * Nn * K1), 256, 0, stream>>>(Z1, (long)B * Nn * K1);
    k_zpad<<<blk((long)B * GH2 * (LDN - Nn)), 256, 0, stream>>>(T1, GH2, (long)GH2 * LDN, Nn, LDN, B * GH2 * (LDN - Nn));
    k_zpad<<<blk((long)B * GH2 * (LDN - Nn)), 256, 0, stream>>>(T2, GH2, (long)GH2 * LDN, Nn, LDN, B * GH2 * (LDN - Nn));
    k_zpad<<<blk((long)B * GH  * (LDN - Nn)), 256, 0, stream>>>(T3, GH,  (long)GH  * LDN, Nn, LDN, B * GH * (LDN - Nn));

    // --- scatter node features (feature-major T0[b][f][n]) ---
    k_scatter<<<blk((long)B * (S - 1) * 16), 256, 0, stream>>>(x, T0, B, S, Fp1, LDN, (long)16 * LDN);

    // --- GCN stack: propagate uses Zᵀ = Xᵀ·Ahat (Ahat symmetric) ---
    // L1 propagate: A=T0[16 x 2016], B=Ahat[N=2000][K=2016] -> Z1 row-major [m][32]
    wmma_gemm<<<dim3(1, 32, B), 32, 0, stream>>>(T0, LDN, (long)16 * LDN,
                                                 Ahat, LDN, 0,
                                                 Z1, K1, (long)Nn * K1,
                                                 16, Nn, LDN, nullptr, 0);
    // L1 weight: A=Z1[2000 x 32], B=W1T[256][32] -> T1 feature-major [f][n], relu+bias
    wmma_gemm<<<dim3(32, 4, B), 32, 0, stream>>>(Z1, K1, (long)Nn * K1,
                                                 W1T, K1, 0,
                                                 T1, LDN, (long)GH2 * LDN,
                                                 Nn, GH2, K1, b_gc1, 1);
    // L2 propagate: A=T1[256 x 2016], B=Ahat -> Z2 [m][256]
    wmma_gemm<<<dim3(4, 32, B), 32, 0, stream>>>(T1, LDN, (long)GH2 * LDN,
                                                 Ahat, LDN, 0,
                                                 Z23, GH2, (long)Nn * GH2,
                                                 GH2, Nn, LDN, nullptr, 0);
    // L2 weight: A=Z2[2000 x 256], B=W2T[256][256] -> T2 [f][n], relu+bias
    wmma_gemm<<<dim3(32, 4, B), 32, 0, stream>>>(Z23, GH2, (long)Nn * GH2,
                                                 W2T, GH2, 0,
                                                 T2, LDN, (long)GH2 * LDN,
                                                 Nn, GH2, GH2, b_gc2, 1);
    // L3 propagate: A=T2 -> Z3 [m][256]
    wmma_gemm<<<dim3(4, 32, B), 32, 0, stream>>>(T2, LDN, (long)GH2 * LDN,
                                                 Ahat, LDN, 0,
                                                 Z23, GH2, (long)Nn * GH2,
                                                 GH2, Nn, LDN, nullptr, 0);
    // L3 weight: A=Z3[2000 x 256], B=W3T[512][256] -> T3 [f][n], relu+bias
    wmma_gemm<<<dim3(32, 8, B), 32, 0, stream>>>(Z23, GH2, (long)Nn * GH2,
                                                 W3T, GH2, 0,
                                                 T3, LDN, (long)GH * LDN,
                                                 Nn, GH, GH2, b_gc3, 1);

    // --- gather + MLP head (f32 scalar, tiny) ---
    k_gather<<<blk((long)B * GH), 256, 0, stream>>>(x, T3, gsel, B, S, Fp1, GH, LDN, (long)GH * LDN);
    k_fc0   <<<blk((long)B * FH), 256, 0, stream>>>(x, w_fc0, b_fc0, h0p, B, S, Fp1, FH);
    k_bn_leaky<<<blk(FH), 256, 0, stream>>>(h0p, g_fc0, be_fc0, h0, B, FH);
    k_fc1   <<<blk((long)B * FH), 256, 0, stream>>>(h0, gsel, w_fc1, b_fc1, h1p, B, FH, GH, FH);
    k_bn_leaky<<<blk(FH), 256, 0, stream>>>(h1p, g_fc1, be_fc1, h1, B, FH);
    k_head  <<<blk(B), 256, 0, stream>>>(h1, w_fc2, b_fc2, (float*)d_out, B, FH);
}
